// BoxEncoder_27152783245664
// MI455X (gfx1250) — matrix operand order repaired and checked by reading, not compiled, not run on hardware
//
#include <hip/hip_runtime.h>
#include <hip/hip_bf16.h>
#include <math.h>

// Shapes: B=128, T=200, NC=3, NB=2, D=512, DH=256
// out: (128, 1800, 512) f32;  rows 0..599 = dist tokens, 600..1799 = box tokens

typedef __attribute__((ext_vector_type(16))) _Float16 v16h;
typedef __attribute__((ext_vector_type(8)))  float    v8f;

union V16H { v16h v; uint4 u[2]; };

struct Meta {
    int   sel_off;   // (cat*3 + cam) * 512 -> offset into base table
    float conf, cx, cy;
    int   out_off;   // element offset of this token row in out
    float present;   // 1.0 present / 0.0 missing
};

// ---------------------------------------------------------------------------
// Prep: w2 (256x512 f32) -> f16, and base[cat][cam][n] table (9x512)
// ---------------------------------------------------------------------------
__global__ void box_encoder_prep(const float* __restrict__ w2,
                                 const float* __restrict__ cat_t,
                                 const float* __restrict__ cam_t,
                                 const float* __restrict__ b2,
                                 const float* __restrict__ confb,
                                 const float* __restrict__ centerb,
                                 _Float16* __restrict__ w2h,
                                 float* __restrict__ basew) {
    int idx = blockIdx.x * blockDim.x + threadIdx.x;
    if (idx < 256 * 512) {
        w2h[idx] = (_Float16)w2[idx];
    } else {
        int e = idx - 256 * 512;
        if (e < 9 * 512) {
            int n  = e & 511;
            int cc = e >> 9;          // 0..8
            int cat = cc / 3, cam = cc % 3;
            basew[e] = b2[n] + confb[n] + centerb[n]
                     + cat_t[cat * 512 + n] + cam_t[cam * 512 + n];
        }
    }
}

// ---------------------------------------------------------------------------
// Main fused kernel: one block = 64 boxes (32 camera pairs), 128 threads
// Phase 2: 4 waves x (8 N-tiles of 16 cols); each B fragment reused by 4 m-tiles
// __launch_bounds__(128, 3): ~340 VGPR budget -> 3 waves/SIMD without the
// scratch spills seen at the 256-VGPR cap, and without the full 32-A-fragment
// hoist (580 VGPRs, 1 wave/SIMD) the unconstrained compiler picks.
// ---------------------------------------------------------------------------
__global__ void __launch_bounds__(128, 3)
box_encoder_main(const float* __restrict__ box,
                 const float* __restrict__ w1,
                 const float* __restrict__ b1,
                 const float* __restrict__ lng,
                 const float* __restrict__ lnb,
                 const float* __restrict__ confw,
                 const float* __restrict__ centerw,
                 const float* __restrict__ missing,
                 const float* __restrict__ distw,
                 const float* __restrict__ distb,
                 const float* __restrict__ scalep,
                 const _Float16* __restrict__ w2h,
                 const float* __restrict__ basew,
                 float* __restrict__ out) {
    __shared__ float s_w1[10 * 256];
    __shared__ float s_b1[256], s_lng[256], s_lnb[256];
    __shared__ float s_confw[512], s_cw0[512], s_cw1[512], s_missing[512];
    __shared__ __align__(16) _Float16 s_hA[64 * 256];   // 32 KB
    __shared__ Meta  s_meta[64];

    const int tid  = threadIdx.x;
    const int lane = tid & 31;
    const int wave = tid >> 5;          // 0..3
    const int sub  = tid >> 4;          // half-wave id 0..7
    const int sl   = tid & 15;          // lane within half-wave

    // ---- stage weights into LDS ----
    for (int i = tid; i < 10 * 256; i += 128) s_w1[i] = w1[i];
    for (int i = tid; i < 256; i += 128) {
        s_b1[i] = b1[i]; s_lng[i] = lng[i]; s_lnb[i] = lnb[i];
    }
    for (int i = tid; i < 512; i += 128) {
        s_confw[i]   = confw[i];
        s_cw0[i]     = centerw[i];         // center_w[0][n]
        s_cw1[i]     = centerw[512 + i];   // center_w[1][n]
        s_missing[i] = missing[i];
    }
    __syncthreads();

    const float inx = 1.0f / 640.0f, iny = 1.0f / 400.0f;

    // ---- phase 1: per-box features, 10->256 matmul, LN, GELU -> f16 in LDS ----
    for (int r = 0; r < 8; ++r) {
        const int bx = r * 8 + sub;          // box within block: 0..63
        const int g  = bx >> 1;              // pair within block: 0..31
        const int nb = bx & 1;
        const int G  = blockIdx.x * 32 + g;  // global (b,t,cam) group, < 76800
        const int bb  = G / 600;
        const int rem = G - bb * 600;        // t*3 + cam
        const int t   = rem / 3;
        const int cam = rem - t * 3;

        const float* bp = box + (((size_t)(bb * 200 + t)) * 6 + cam * 2) * 6;
        float pA[6], pQ[6];
        #pragma unroll
        for (int f = 0; f < 6; ++f) { pA[f] = bp[f]; pQ[f] = bp[6 + f]; }

        float ax1 = pA[0] * inx, ay1 = pA[1] * iny, ax2 = pA[2] * inx, ay2 = pA[3] * iny;
        float qx1 = pQ[0] * inx, qy1 = pQ[1] * iny, qx2 = pQ[2] * inx, qy2 = pQ[3] * iny;
        float presA = ((ax1 + ay1 + ax2 + ay2) != 0.0f) ? 1.0f : 0.0f;
        float presQ = ((qx1 + qy1 + qx2 + qy2) != 0.0f) ? 1.0f : 0.0f;
        float keyA = pA[4] + (1.0f - presA) * 1000.0f;
        float keyQ = pQ[4] + (1.0f - presQ) * 1000.0f;
        const bool swapped   = keyQ < keyA;               // stable 2-sort
        const bool takeSecnd = (nb == 0) ? swapped : !swapped;

        float sx1 = takeSecnd ? qx1 : ax1, sy1 = takeSecnd ? qy1 : ay1;
        float sx2 = takeSecnd ? qx2 : ax2, sy2 = takeSecnd ? qy2 : ay2;
        float scat  = takeSecnd ? pQ[4] : pA[4];
        float sconf = takeSecnd ? pQ[5] : pA[5];

        float w  = sx2 - sx1, h = sy2 - sy1;
        float cx = 0.5f * (sx1 + sx2), cy = 0.5f * (sy1 + sy2);
        float area = w * h, aspect = w / (h + 1e-6f);
        float gf[10] = { sx1, sy1, sx2, sy2, w, h, cx, cy, area, aspect };

        // 10 -> 256 matmul: lane handles j = sl + 16*i
        float hv[16];
        #pragma unroll
        for (int i = 0; i < 16; ++i) {
            int j = sl + i * 16;
            float acc = s_b1[j];
            #pragma unroll
            for (int f = 0; f < 10; ++f) acc += gf[f] * s_w1[f * 256 + j];
            hv[i] = acc;
        }
        // LayerNorm over 256 (16 lanes x 16 values), half-wave shuffles
        float s = 0.0f;
        #pragma unroll
        for (int i = 0; i < 16; ++i) s += hv[i];
        s += __shfl_xor(s, 1, 32); s += __shfl_xor(s, 2, 32);
        s += __shfl_xor(s, 4, 32); s += __shfl_xor(s, 8, 32);
        float mu = s * (1.0f / 256.0f);
        float vs = 0.0f;
        #pragma unroll
        for (int i = 0; i < 16; ++i) { float d = hv[i] - mu; vs += d * d; }
        vs += __shfl_xor(vs, 1, 32); vs += __shfl_xor(vs, 2, 32);
        vs += __shfl_xor(vs, 4, 32); vs += __shfl_xor(vs, 8, 32);
        float rstd = rsqrtf(vs * (1.0f / 256.0f) + 1e-5f);
        #pragma unroll
        for (int i = 0; i < 16; ++i) {
            int j = sl + i * 16;
            float y  = (hv[i] - mu) * rstd * s_lng[j] + s_lnb[j];
            float ge = 0.5f * y * (1.0f + erff(y * 0.70710678118654752f));
            s_hA[bx * 256 + j] = (_Float16)ge;
        }

        if (sl == 0) {
            int cat = (int)scat; cat = cat < 0 ? 0 : (cat > 2 ? 2 : cat);
            Meta mm;
            mm.sel_off = (cat * 3 + cam) * 512;
            mm.conf    = sconf;
            mm.cx      = cx;
            mm.cy      = cy;
            mm.out_off = (bb * 1800 + 600 + rem * 2 + nb) * 512;
            mm.present = ((sx1 + sy1 + sx2 + sy2) != 0.0f) ? 1.0f : 0.0f;
            s_meta[bx] = mm;
        }

        // dist token for this camera pair (done by the nb==0 half-wave)
        if (nb == 0) {
            float ox1 = takeSecnd ? ax1 : qx1, oy1 = takeSecnd ? ay1 : qy1;
            float ox2 = takeSecnd ? ax2 : qx2, oy2 = takeSecnd ? ay2 : qy2;
            float bcx = 0.5f * (ox1 + ox2), bcy = 0.5f * (oy1 + oy2);
            float dx = cx - bcx, dy = cy - bcy;
            float dist = sqrtf(dx * dx + dy * dy);
            int dbase = (bb * 1800 + rem) * 512;
            #pragma unroll
            for (int i2 = 0; i2 < 32; ++i2) {
                int n = sl + i2 * 16;
                out[dbase + n] = dist * distw[n] + distb[n];
            }
        }
    }
    __syncthreads();

    // ---- phase 2: WMMA GEMM  H(64x256 f16, LDS) x W2(256x512 f16) ----
    // Each wave: 8 N-tiles (128 cols); each B fragment feeds 4 m-tiles.
    const float scalev = scalep[0];
    const int m  = lane & 15;
    const int hi = lane >> 4;   // 0: lanes 0-15, 1: lanes 16-31

    #pragma unroll 1
    for (int q = 0; q < 8; ++q) {
        const int n0 = (wave * 8 + q) * 16;
        v8f acc[4];
        #pragma unroll
        for (int mt = 0; mt < 4; ++mt) acc[mt] = (v8f){0.f,0.f,0.f,0.f,0.f,0.f,0.f,0.f};

        // unroll-2: two-stage pipeline (B load of next step overlaps WMMAs)
        // without tempting the compiler into hoisting all 32 A fragments.
        #pragma unroll 2
        for (int kk = 0; kk < 8; ++kk) {
            // B fragment: lane L holds col n0 + (L&15), K in same order as A
            const int krow = kk * 32 + hi * 8;
            const _Float16* bcol = w2h + (size_t)krow * 512 + n0 + m;
            V16H bf;
            #pragma unroll
            for (int s2 = 0; s2 < 8; ++s2) {
                bf.v[s2]     = bcol[s2 * 512];
                bf.v[8 + s2] = bcol[(16 + s2) * 512];
            }
            // Batch all 4 A fragments first (8 independent ds_load_b128 the
            // scheduler can clause + partially wait on), then the 4 WMMAs.
            //  A layout (16-bit 16x32):
            //  lane<16: halves 0..7 = K k0+0..7,  8..15 = K k0+16..23
            //  lane>=16: halves 0..7 = K k0+8..15, 8..15 = K k0+24..31
            V16H af[4];
            #pragma unroll
            for (int mt = 0; mt < 4; ++mt) {
                const int row = mt * 16 + m;
                af[mt].u[0] = *(const uint4*)&s_hA[row * 256 + kk * 32 + hi * 8];
                af[mt].u[1] = *(const uint4*)&s_hA[row * 256 + kk * 32 + 16 + hi * 8];
            }
            #pragma unroll
            for (int mt = 0; mt < 4; ++mt) {
                acc[mt] = __builtin_amdgcn_wmma_f32_16x16x32_f16(
                    false, af[mt].v, false, bf.v, (short)0, acc[mt], false, false);
            }
        }

        // epilogue + store; C/D layout: lane<16 -> M=r, lane>=16 -> M=r+8, N=lane&15
        const int n = n0 + m;
        const float cfn  = s_confw[n];
        const float cw0n = s_cw0[n];
        const float cw1n = s_cw1[n];
        const float msn  = s_missing[n];
        #pragma unroll
        for (int mt = 0; mt < 4; ++mt) {
            #pragma unroll
            for (int rr = 0; rr < 8; ++rr) {
                const int M = mt * 16 + rr + hi * 8;
                Meta mm = s_meta[M];
                float tok = (acc[mt][rr] + basew[mm.sel_off + n]
                             + mm.conf * cfn + mm.cx * cw0n + mm.cy * cw1n) * scalev;
                // branchless missing-mask select
                out[(size_t)mm.out_off + n] = msn + mm.present * (tok - msn);
            }
        }
    }
}

// ---------------------------------------------------------------------------
extern "C" void kernel_launch(void* const* d_in, const int* in_sizes, int n_in,
                              void* d_out, int out_size, void* d_ws, size_t ws_size,
                              hipStream_t stream) {
    const float* box      = (const float*)d_in[0];
    const float* cat_tab  = (const float*)d_in[1];
    const float* w1       = (const float*)d_in[2];
    const float* b1       = (const float*)d_in[3];
    const float* lng      = (const float*)d_in[4];
    const float* lnb      = (const float*)d_in[5];
    const float* w2       = (const float*)d_in[6];
    const float* b2       = (const float*)d_in[7];
    const float* confw    = (const float*)d_in[8];
    const float* confb    = (const float*)d_in[9];
    const float* centerw  = (const float*)d_in[10];
    const float* centerb  = (const float*)d_in[11];
    const float* missing  = (const float*)d_in[12];
    const float* distw    = (const float*)d_in[13];
    const float* distb    = (const float*)d_in[14];
    const float* cam_tab  = (const float*)d_in[15];
    const float* scalep   = (const float*)d_in[16];

    _Float16* w2h   = (_Float16*)d_ws;                       // 256*512*2 = 262144 B
    float*    basew = (float*)((char*)d_ws + 262144);        // 9*512*4  = 18432 B

    // prep: 131072 + 4608 elements
    {
        int total = 256 * 512 + 9 * 512;
        int blocks = (total + 255) / 256;
        box_encoder_prep<<<blocks, 256, 0, stream>>>(
            w2, cat_tab, cam_tab, b2, confb, centerb, w2h, basew);
    }
    // main: 153600 boxes / 64 per block = 2400 blocks
    box_encoder_main<<<2400, 128, 0, stream>>>(
        box, w1, b1, lng, lnb, confw, centerw, missing, distw, distb,
        scalep, w2h, basew, (float*)d_out);
}
